// MultiBoxLoss_43550968381671
// MI455X (gfx1250) — compile-verified, matrix operand order checked by reading, not run on hardware
//
#include <hip/hip_runtime.h>

typedef __attribute__((ext_vector_type(16))) _Float16 v16h;
typedef __attribute__((ext_vector_type(8)))  float    v8f;

namespace {
constexpr int B_ = 8, P_ = 19248, C_ = 81, G_ = 16, K_ = 32, HM_ = 138;
constexpr int NPIX = HM_ * HM_;            // 19044
constexpr int M_ = 100;                    // masks_to_train
constexpr int MT_ = (M_ + 15) / 16;        // 7 mask tiles
constexpr int PT_ = (NPIX + 15) / 16;      // 1191 pixel tiles
constexpr int TPW = 16;                    // pixel tiles per wave
constexpr int OHBINS = 1024, MHBINS = 512;
constexpr float POS_T = 0.5f, NEG_T = 0.4f;
constexpr float BBOX_ALPHA = 1.5f, CONF_ALPHA = 1.0f, MASK_ALPHA = 6.125f;
constexpr float EPSV = 1e-7f;

// workspace layout (4-byte words); zeroed region first
constexpr int OFF_NUMPOS  = 0;                       // [B] int
constexpr int OFF_TOTPOS  = OFF_NUMPOS + B_;         // [1] int
constexpr int OFF_SLSUM   = OFF_TOTPOS + 1;          // [1] float
constexpr int OFF_CESUM   = OFF_SLSUM + 1;           // [1] float
constexpr int OFF_OHIST   = OFF_CESUM + 1;           // [B*1024] int
constexpr int OFF_SELVAL  = OFF_OHIST + B_ * OHBINS; // [B*M] int
constexpr int OFF_PERMASK = OFF_SELVAL + B_ * M_;    // [B*M] float
constexpr int ZERO_WORDS  = OFF_PERMASK + B_ * M_;
constexpr int OFF_NEGTH   = ZERO_WORDS;              // [B] float
constexpr int OFF_BTO     = OFF_NEGTH + B_;          // [B*P] float
constexpr int OFF_BTI     = OFF_BTO + B_ * P_;       // [B*P] int
constexpr int OFF_CONFT   = OFF_BTI + B_ * P_;       // [B*P] int
constexpr int OFF_LSE     = OFF_CONFT + B_ * P_;     // [B*P] float
constexpr int OFF_LOSSC   = OFF_LSE + B_ * P_;       // [B*P] float
constexpr int OFF_BPI     = OFF_LOSSC + B_ * P_;     // [B*G] int
constexpr int OFF_SELP    = OFF_BPI + B_ * G_;       // [B*M] int
constexpr int OFF_SELG    = OFF_SELP + B_ * M_;      // [B*M] int
constexpr int OFF_SELBOX  = OFF_SELG + B_ * M_;      // [B*M*4] float
} // namespace

__global__ void k_init(int* ws_i) {
  int i = blockIdx.x * 256 + threadIdx.x;
  if (i < ZERO_WORDS) ws_i[i] = 0;
}

// best GT per prior: bto/bti
__global__ void k_match(const float* __restrict__ priors,
                        const float* __restrict__ truths,
                        float* ws_f, int* ws_i) {
  __shared__ float tsh[G_ * 4];
  int b = blockIdx.y;
  int p = blockIdx.x * 256 + threadIdx.x;
  if (threadIdx.x < G_ * 4) tsh[threadIdx.x] = truths[b * G_ * 4 + threadIdx.x];
  __syncthreads();
  if (p >= P_) return;
  float cx = priors[p * 4 + 0], cy = priors[p * 4 + 1];
  float w  = priors[p * 4 + 2], h  = priors[p * 4 + 3];
  float bx1 = cx - 0.5f * w, by1 = cy - 0.5f * h;
  float bx2 = cx + 0.5f * w, by2 = cy + 0.5f * h;
  float area_b = (bx2 - bx1) * (by2 - by1);
  float best = -1.f; int besti = 0;
  for (int g = 0; g < G_; ++g) {
    float ax1 = tsh[g * 4 + 0], ay1 = tsh[g * 4 + 1];
    float ax2 = tsh[g * 4 + 2], ay2 = tsh[g * 4 + 3];
    float iw = fmaxf(fminf(ax2, bx2) - fmaxf(ax1, bx1), 0.f);
    float ih = fmaxf(fminf(ay2, by2) - fmaxf(ay1, by1), 0.f);
    float inter = iw * ih;
    float area_a = (ax2 - ax1) * (ay2 - ay1);
    float iou = inter / (area_a + area_b - inter);
    if (iou > best) { best = iou; besti = g; }
  }
  ws_f[OFF_BTO + b * P_ + p] = best;
  ws_i[OFF_BTI + b * P_ + p] = besti;
}

// best prior per GT (argmax over P)
__global__ void k_best_prior(const float* __restrict__ priors,
                             const float* __restrict__ truths, int* ws_i) {
  int b = blockIdx.y, g = blockIdx.x, tid = threadIdx.x;
  float ax1 = truths[(b * G_ + g) * 4 + 0], ay1 = truths[(b * G_ + g) * 4 + 1];
  float ax2 = truths[(b * G_ + g) * 4 + 2], ay2 = truths[(b * G_ + g) * 4 + 3];
  float area_a = (ax2 - ax1) * (ay2 - ay1);
  float best = -1.f; int besti = 0;
  for (int p = tid; p < P_; p += 256) {
    float cx = priors[p * 4 + 0], cy = priors[p * 4 + 1];
    float w  = priors[p * 4 + 2], h  = priors[p * 4 + 3];
    float bx1 = cx - 0.5f * w, by1 = cy - 0.5f * h;
    float bx2 = cx + 0.5f * w, by2 = cy + 0.5f * h;
    float iw = fmaxf(fminf(ax2, bx2) - fmaxf(ax1, bx1), 0.f);
    float ih = fmaxf(fminf(ay2, by2) - fmaxf(ay1, by1), 0.f);
    float inter = iw * ih;
    float iou = inter / (area_a + (bx2 - bx1) * (by2 - by1) - inter);
    if (iou > best || (iou == best && p < besti)) { best = iou; besti = p; }
  }
  __shared__ float sv[256]; __shared__ int si[256];
  sv[tid] = best; si[tid] = besti; __syncthreads();
  for (int s = 128; s > 0; s >>= 1) {
    if (tid < s) {
      if (sv[tid + s] > sv[tid] || (sv[tid + s] == sv[tid] && si[tid + s] < si[tid])) {
        sv[tid] = sv[tid + s]; si[tid] = si[tid + s];
      }
    }
    __syncthreads();
  }
  if (tid == 0) ws_i[OFF_BPI + b * G_ + g] = si[0];
}

__global__ void k_force(float* ws_f, int* ws_i) {
  int b = blockIdx.x;
  if (threadIdx.x != 0) return;
  for (int g = 0; g < G_; ++g) {
    int p = ws_i[OFF_BPI + b * G_ + g];
    ws_f[OFF_BTO + b * P_ + p] = 2.0f;
    ws_i[OFF_BTI + b * P_ + p] = g;
  }
}

// conf_t, smooth-L1, LSE + OHEM histogram
__global__ void k_conf_loc(const float* __restrict__ loc_data,
                           const float* __restrict__ conf_data,
                           const float* __restrict__ priors,
                           const float* __restrict__ truths,
                           const int* __restrict__ labels,
                           float* ws_f, int* ws_i) {
  int b = blockIdx.y, tid = threadIdx.x;
  int p = blockIdx.x * 256 + tid;
  float sl_local = 0.f; int pos_local = 0;
  if (p < P_) {
    float ov = ws_f[OFF_BTO + b * P_ + p];
    int g = ws_i[OFF_BTI + b * P_ + p];
    int conf = labels[b * G_ + g] + 1;
    if (ov < POS_T) conf = -1;
    if (ov < NEG_T) conf = 0;
    ws_i[OFF_CONFT + b * P_ + p] = conf;
    bool pos = conf > 0;
    pos_local = pos ? 1 : 0;
    float posf = pos ? 1.f : 0.f;
    float m0 = truths[(b * G_ + g) * 4 + 0], m1 = truths[(b * G_ + g) * 4 + 1];
    float m2 = truths[(b * G_ + g) * 4 + 2], m3 = truths[(b * G_ + g) * 4 + 3];
    float pcx = priors[p * 4 + 0], pcy = priors[p * 4 + 1];
    float pw  = priors[p * 4 + 2], ph  = priors[p * 4 + 3];
    float lt[4];
    lt[0] = ((m0 + m2) * 0.5f - pcx) / (0.1f * pw);
    lt[1] = ((m1 + m3) * 0.5f - pcy) / (0.1f * ph);
    lt[2] = __logf(fmaxf(m2 - m0, EPSV) / pw) * 5.f;
    lt[3] = __logf(fmaxf(m3 - m1, EPSV) / ph) * 5.f;
#pragma unroll
    for (int i = 0; i < 4; ++i) {
      float d = loc_data[((size_t)b * P_ + p) * 4 + i] - lt[i];
      float ad = fabsf(d);
      sl_local += ((ad < 1.f) ? 0.5f * d * d : ad - 0.5f) * posf;
    }
    // branchless two-pass logsumexp over C=81
    const float* cp = conf_data + ((size_t)b * P_ + p) * C_;
    float c0 = cp[0], mx = cp[0];
    for (int c = 1; c < C_; ++c) mx = fmaxf(mx, cp[c]);
    float s = 0.f;
    for (int c = 0; c < C_; ++c) s += __expf(cp[c] - mx);
    float lse = mx + __logf(s);
    ws_f[OFF_LSE + b * P_ + p] = lse;
    float lc = (conf == 0) ? (lse - c0) : 0.f;
    ws_f[OFF_LOSSC + b * P_ + p] = lc;
    if (conf == 0) {
      int bin = (int)(lc * 64.f);
      bin = max(0, min(OHBINS - 1, bin));
      atomicAdd(&ws_i[OFF_OHIST + b * OHBINS + bin], 1);
    }
  }
  __shared__ float ssl[256]; __shared__ int spc[256];
  ssl[tid] = sl_local; spc[tid] = pos_local; __syncthreads();
  for (int s = 128; s > 0; s >>= 1) {
    if (tid < s) { ssl[tid] += ssl[tid + s]; spc[tid] += spc[tid + s]; }
    __syncthreads();
  }
  if (tid == 0) {
    atomicAdd(&ws_f[OFF_SLSUM], ssl[0]);
    atomicAdd(&ws_i[OFF_NUMPOS + b], spc[0]);
    atomicAdd(&ws_i[OFF_TOTPOS], spc[0]);
  }
}

__global__ void k_ohem_thresh(float* ws_f, int* ws_i) {
  int b = blockIdx.x;
  if (threadIdx.x != 0) return;
  int nn = min(3 * ws_i[OFF_NUMPOS + b], P_ - 1);
  int cum = 0; float th = 1e30f;
  for (int bin = OHBINS - 1; bin >= 0; --bin) {
    cum += ws_i[OFF_OHIST + b * OHBINS + bin];
    if (cum >= nn) { th = bin * (1.f / 64.f); break; }
  }
  ws_f[OFF_NEGTH + b] = th;
}

__global__ void k_conf_loss(const float* __restrict__ conf_data,
                            float* ws_f, int* ws_i) {
  int b = blockIdx.y, tid = threadIdx.x;
  int p = blockIdx.x * 256 + tid;
  float ce_local = 0.f;
  if (p < P_) {
    int conf = ws_i[OFF_CONFT + b * P_ + p];
    bool pos = conf > 0;
    bool neg = (conf == 0) && (ws_f[OFF_LOSSC + b * P_ + p] > ws_f[OFF_NEGTH + b]);
    if (pos || neg) {
      int tgt = max(conf, 0);
      ce_local = ws_f[OFF_LSE + b * P_ + p] -
                 conf_data[((size_t)b * P_ + p) * C_ + tgt];
    }
  }
  __shared__ float red[256];
  red[tid] = ce_local; __syncthreads();
  for (int s = 128; s > 0; s >>= 1) {
    if (tid < s) red[tid] += red[tid + s];
    __syncthreads();
  }
  if (tid == 0) atomicAdd(&ws_f[OFF_CESUM], red[0]);
}

// top-~100 positives per image via histogram threshold + compaction
__global__ void k_mask_select(const float* __restrict__ truths,
                              float* ws_f, int* ws_i) {
  int b = blockIdx.x, tid = threadIdx.x;
  __shared__ int hist[MHBINS];
  __shared__ int thbin, cnt;
  for (int i = tid; i < MHBINS; i += 256) hist[i] = 0;
  if (tid == 0) cnt = 0;
  __syncthreads();
  for (int p = tid; p < P_; p += 256) {
    if (ws_i[OFF_CONFT + b * P_ + p] > 0) {
      float ov = ws_f[OFF_BTO + b * P_ + p];
      int bin = min(MHBINS - 1, max(0, (int)(ov * 256.f)));
      atomicAdd(&hist[bin], 1);
    }
  }
  __syncthreads();
  if (tid == 0) {
    int cum = 0, tb = 0;
    for (int bin = MHBINS - 1; bin >= 0; --bin) {
      cum += hist[bin];
      if (cum >= M_) { tb = bin; break; }
    }
    thbin = (cum >= M_) ? tb : 0;
  }
  __syncthreads();
  for (int p = tid; p < P_; p += 256) {
    if (ws_i[OFF_CONFT + b * P_ + p] > 0) {
      float ov = ws_f[OFF_BTO + b * P_ + p];
      int bin = min(MHBINS - 1, max(0, (int)(ov * 256.f)));
      if (bin >= thbin) {
        int slot = atomicAdd(&cnt, 1);
        if (slot < M_) {
          int g = ws_i[OFF_BTI + b * P_ + p];
          ws_i[OFF_SELP + b * M_ + slot] = p;
          ws_i[OFF_SELG + b * M_ + slot] = g;
          ws_i[OFF_SELVAL + b * M_ + slot] = 1;
          float m0 = truths[(b * G_ + g) * 4 + 0], m1 = truths[(b * G_ + g) * 4 + 1];
          float m2 = truths[(b * G_ + g) * 4 + 2], m3 = truths[(b * G_ + g) * 4 + 3];
          float* bx = &ws_f[OFF_SELBOX + (b * M_ + slot) * 4];
          bx[0] = fmaxf(fminf(m0, m2) * HM_ - 1.f, 0.f);
          bx[1] = fminf(fmaxf(m0, m2) * HM_ + 1.f, (float)HM_);
          bx[2] = fmaxf(fminf(m1, m3) * HM_ - 1.f, 0.f);
          bx[3] = fminf(fmaxf(m1, m3) * HM_ + 1.f, (float)HM_);
        }
      }
    }
  }
}

// mask lincomb GEMM via v_wmma_f32_16x16x32_f16 + sigmoid/crop/BCE.
// One wave owns one 16-mask column tile and sweeps TPW pixel tiles,
// fully branchless (EXEC stays all-ones for every WMMA).
__global__ void __launch_bounds__(32)
k_mask_wmma(const float* __restrict__ proto, const float* __restrict__ mask_coef,
            const float* __restrict__ gt_masks, float* ws_f, int* ws_i) {
  int b = blockIdx.z, mtile = blockIdx.y;
  int tile0 = blockIdx.x * TPW;
  int lane = threadIdx.x;
  int group = lane >> 4;          // 0: K 0..7/16..23, 1: K 8..15/24..31
  int lcol = lane & 15;
  int k0 = group * 8;

  // --- B fragment: coeff column (mask), loaded once ---
  int m = mtile * 16 + lcol;
  int mm = min(m, M_ - 1);
  int valid = (m < M_) ? ws_i[OFF_SELVAL + b * M_ + mm] : 0;
  int psel = ws_i[OFF_SELP + b * M_ + mm];
  psel = valid ? min(max(psel, 0), P_ - 1) : 0;
  float vf = valid ? 1.f : 0.f;
  const float* cb = mask_coef + ((size_t)b * P_ + psel) * K_;
  float4 q0 = *(const float4*)(cb + k0);
  float4 q1 = *(const float4*)(cb + k0 + 4);
  float4 q2 = *(const float4*)(cb + 16 + k0);
  float4 q3 = *(const float4*)(cb + 16 + k0 + 4);
  v16h Bf;
  Bf[0]=(_Float16)(q0.x*vf); Bf[1]=(_Float16)(q0.y*vf); Bf[2]=(_Float16)(q0.z*vf); Bf[3]=(_Float16)(q0.w*vf);
  Bf[4]=(_Float16)(q1.x*vf); Bf[5]=(_Float16)(q1.y*vf); Bf[6]=(_Float16)(q1.z*vf); Bf[7]=(_Float16)(q1.w*vf);
  Bf[8]=(_Float16)(q2.x*vf); Bf[9]=(_Float16)(q2.y*vf); Bf[10]=(_Float16)(q2.z*vf); Bf[11]=(_Float16)(q2.w*vf);
  Bf[12]=(_Float16)(q3.x*vf); Bf[13]=(_Float16)(q3.y*vf); Bf[14]=(_Float16)(q3.z*vf); Bf[15]=(_Float16)(q3.w*vf);

  // --- crop box + gt index, loaded once (clamped, gated later) ---
  const float* bx = ws_f + OFF_SELBOX + (size_t)(b * M_ + mm) * 4;
  float x1 = bx[0], x2 = bx[1], y1 = bx[2], y2 = bx[3];
  int gidx = ws_i[OFF_SELG + b * M_ + mm];
  gidx = valid ? min(max(gidx, 0), G_ - 1) : 0;
  const float* gm = gt_masks + ((size_t)b * G_ + gidx) * NPIX;
  const float* pb = proto + (size_t)b * NPIX * K_;

  float lane_sum = 0.f;
#pragma unroll
  for (int t = 0; t < TPW; ++t) {
    int ptile = min(tile0 + t, PT_ - 1);
    float act = (tile0 + t < PT_) ? 1.f : 0.f;
    // A fragment: 16 proto pixels x 32 K
    int pixA = min(ptile * 16 + lcol, NPIX - 1);
    const float* ab = pb + (size_t)pixA * K_;
    float4 r0 = *(const float4*)(ab + k0);
    float4 r1 = *(const float4*)(ab + k0 + 4);
    float4 r2 = *(const float4*)(ab + 16 + k0);
    float4 r3 = *(const float4*)(ab + 16 + k0 + 4);
    v16h Af;
    Af[0]=(_Float16)r0.x; Af[1]=(_Float16)r0.y; Af[2]=(_Float16)r0.z; Af[3]=(_Float16)r0.w;
    Af[4]=(_Float16)r1.x; Af[5]=(_Float16)r1.y; Af[6]=(_Float16)r1.z; Af[7]=(_Float16)r1.w;
    Af[8]=(_Float16)r2.x; Af[9]=(_Float16)r2.y; Af[10]=(_Float16)r2.z; Af[11]=(_Float16)r2.w;
    Af[12]=(_Float16)r3.x; Af[13]=(_Float16)r3.y; Af[14]=(_Float16)r3.z; Af[15]=(_Float16)r3.w;

    v8f Cf = {};
    Cf = __builtin_amdgcn_wmma_f32_16x16x32_f16(false, Af, false, Bf,
                                                (short)0, Cf, false, false);

    int base = ptile * 16 + 8 * group;   // D row = pixel-in-tile
#pragma unroll
    for (int v = 0; v < 8; ++v) {
      int pix = base + v;
      int pixc = min(pix, NPIX - 1);
      float inb = (pix < NPIX) ? act : 0.f;
      int h = pixc / HM_;
      int w = pixc - h * HM_;
      float wfp = (float)w, hfp = (float)h;
      float crop = (wfp >= x1 && wfp < x2 && hfp >= y1 && hfp < y2) ? 1.f : 0.f;
      float pr = crop / (1.f + __expf(-Cf[v]));
      pr = fminf(fmaxf(pr, EPSV), 1.f - EPSV);
      float mt = gm[pixc];
      float bce = -(mt * __logf(pr) + (1.f - mt) * __logf(1.f - pr));
      lane_sum += bce * inb;
    }
  }
  if (valid) atomicAdd(&ws_f[OFF_PERMASK + b * M_ + mm], lane_sum);
}

__global__ void k_final(float* ws_f, int* ws_i, float* out) {
  int tid = threadIdx.x;
  __shared__ float red[256];
  float local = 0.f;
  for (int i = tid; i < B_ * M_; i += 256) {
    if (ws_i[OFF_SELVAL + i]) {
      const float* bx = ws_f + OFF_SELBOX + (size_t)i * 4;
      float area = fmaxf((bx[1] - bx[0]) * (bx[3] - bx[2]), 1.f);
      local += ws_f[OFF_PERMASK + i] / area;
    }
  }
  red[tid] = local; __syncthreads();
  for (int s = 128; s > 0; s >>= 1) {
    if (tid < s) red[tid] += red[tid + s];
    __syncthreads();
  }
  if (tid == 0) {
    float tp = (float)max(ws_i[OFF_TOTPOS], 1);
    out[0] = ws_f[OFF_SLSUM] * BBOX_ALPHA / tp;
    out[1] = ws_f[OFF_CESUM] * CONF_ALPHA / tp;
    out[2] = red[0] * MASK_ALPHA / tp;
  }
}

extern "C" void kernel_launch(void* const* d_in, const int* in_sizes, int n_in,
                              void* d_out, int out_size, void* d_ws, size_t ws_size,
                              hipStream_t stream) {
  (void)in_sizes; (void)n_in; (void)out_size; (void)ws_size;
  const float* loc_data  = (const float*)d_in[0];
  const float* conf_data = (const float*)d_in[1];
  const float* mask_coef = (const float*)d_in[2];
  const float* proto     = (const float*)d_in[3];
  const float* priors    = (const float*)d_in[4];
  const float* truths    = (const float*)d_in[5];
  const float* gt_masks  = (const float*)d_in[6];
  const int*   labels    = (const int*)d_in[7];
  float* ws_f = (float*)d_ws;
  int*   ws_i = (int*)d_ws;
  float* out  = (float*)d_out;

  dim3 bp((P_ + 255) / 256, B_);
  k_init<<<(ZERO_WORDS + 255) / 256, 256, 0, stream>>>(ws_i);
  k_match<<<bp, 256, 0, stream>>>(priors, truths, ws_f, ws_i);
  k_best_prior<<<dim3(G_, B_), 256, 0, stream>>>(priors, truths, ws_i);
  k_force<<<B_, 1, 0, stream>>>(ws_f, ws_i);
  k_conf_loc<<<bp, 256, 0, stream>>>(loc_data, conf_data, priors, truths, labels, ws_f, ws_i);
  k_ohem_thresh<<<B_, 1, 0, stream>>>(ws_f, ws_i);
  k_conf_loss<<<bp, 256, 0, stream>>>(conf_data, ws_f, ws_i);
  k_mask_select<<<B_, 256, 0, stream>>>(truths, ws_f, ws_i);
  k_mask_wmma<<<dim3((PT_ + TPW - 1) / TPW, MT_, B_), 32, 0, stream>>>(
      proto, mask_coef, gt_masks, ws_f, ws_i);
  k_final<<<1, 256, 0, stream>>>(ws_f, ws_i, out);
}
